// CHAN_29764123361280
// MI455X (gfx1250) — matrix operand order repaired and checked by reading, not compiled, not run on hardware
//
#include <hip/hip_runtime.h>
#include <hip/hip_bf16.h>
#include <math.h>

// ---------------------------------------------------------------------------
// Sizes from the reference
// ---------------------------------------------------------------------------
#define Bb    8
#define Ss    32
#define Ll    128
#define CIN   1024
#define C1c   512
#define C2c   256
#define D1d   512
#define D2d   128
#define CDd   300
#define SIMs  128
#define L4    32
#define Nn    256   // B*S

typedef _Float16 v16h __attribute__((ext_vector_type(16)));
typedef _Float16 v8h  __attribute__((ext_vector_type(8)));
typedef float    v8f  __attribute__((ext_vector_type(8)));
typedef float    v4f  __attribute__((ext_vector_type(4)));

// ---------------------------------------------------------------------------
// WMMA helpers (CDNA5: 16x16x32 f16 -> f32 accumulate)
// A 16x32 (MxK): lane l holds row m=l&15; halves 0..7 -> K = hi+j,
//                halves 8..15 -> K = 16+hi+(j-8), hi = (l>>4)*8.
// B 32x16 (KxN): lane l holds column N=l&15; halves as A's K order.
// C 16x16: lane n-col = l&15; vgpr i -> m = i + (l>>4)*8.
// ---------------------------------------------------------------------------
__device__ __forceinline__ v8f wmma16(v16h a, v16h b, v8f c) {
  return __builtin_amdgcn_wmma_f32_16x16x32_f16(false, a, false, b, (short)0, c,
                                                false, false);
}

__device__ __forceinline__ v16h a_frag_f32(const float* __restrict__ row, int k0,
                                           int hi) {
  v4f x0 = *(const v4f*)(row + k0 + hi);
  v4f x1 = *(const v4f*)(row + k0 + hi + 4);
  v4f x2 = *(const v4f*)(row + k0 + hi + 16);
  v4f x3 = *(const v4f*)(row + k0 + hi + 20);
  v16h a;
#pragma unroll
  for (int j = 0; j < 4; ++j) {
    a[j]      = (_Float16)x0[j];
    a[4 + j]  = (_Float16)x1[j];
    a[8 + j]  = (_Float16)x2[j];
    a[12 + j] = (_Float16)x3[j];
  }
  return a;
}

__device__ __forceinline__ v16h a_frag_f16(const _Float16* __restrict__ row, int k0,
                                           int hi) {
  v8h lo = *(const v8h*)(row + k0 + hi);
  v8h hp = *(const v8h*)(row + k0 + hi + 16);
  v16h a;
#pragma unroll
  for (int j = 0; j < 8; ++j) { a[j] = lo[j]; a[8 + j] = hp[j]; }
  return a;
}

__device__ __forceinline__ v16h b_frag(const _Float16* __restrict__ Bm, int ldb,
                                       int k0, int n0, int lane) {
  const _Float16* p = Bm + (size_t)(k0 + (lane >> 4) * 8) * ldb + n0 + (lane & 15);
  v16h b;
#pragma unroll
  for (int j = 0; j < 8; ++j) { b[j] = p[(size_t)j * ldb]; b[8 + j] = p[(size_t)(16 + j) * ldb]; }
  return b;
}

// B fragment from an LDS panel [128 rows][64 cols] f16
__device__ __forceinline__ v16h b_frag_lds(const _Float16* Bm, int k0, int n0,
                                           int lane) {
  const _Float16* p = Bm + (k0 + (lane >> 4) * 8) * 64 + n0 + (lane & 15);
  v16h b;
#pragma unroll
  for (int j = 0; j < 8; ++j) { b[j] = p[j * 64]; b[8 + j] = p[(16 + j) * 64]; }
  return b;
}

// async global->LDS 128b copy (VGLOBAL encoding, ASYNCcnt-tracked)
__device__ __forceinline__ void async_copy_b128(const void* gaddr, void* lds_generic) {
  unsigned ldsa = (unsigned)(size_t)lds_generic;  // low 32 bits = LDS offset
  asm volatile("global_load_async_to_lds_b128 %0, %1, off"
               :: "v"(ldsa), "v"(gaddr)
               : "memory");
}
__device__ __forceinline__ void wait_asynccnt0() {
  asm volatile("s_wait_asynccnt 0" ::: "memory");
}

// ---------------------------------------------------------------------------
// Weight prep kernels
// ---------------------------------------------------------------------------
__global__ void zero_fill(float* __restrict__ z, int n) {
  int i = blockIdx.x * blockDim.x + threadIdx.x;
  if (i < n) z[i] = 0.f;
}

// conv weights [Co][Ci][K] (f32) -> [K][Ci][Co] (f16)
__global__ void xpose_conv_w(const float* __restrict__ w, _Float16* __restrict__ wt,
                             int Co, int Ci, int Kk) {
  size_t total = (size_t)Co * Ci * Kk;
  for (size_t idx = (size_t)blockIdx.x * blockDim.x + threadIdx.x; idx < total;
       idx += (size_t)gridDim.x * blockDim.x) {
    int co = (int)(idx % Co);
    size_t r = idx / Co;
    int ci = (int)(r % Ci);
    int k  = (int)(r / Ci);
    wt[idx] = (_Float16)w[((size_t)co * Ci + ci) * Kk + k];
  }
}

__global__ void cvt_f16_kernel(const float* __restrict__ in,
                               _Float16* __restrict__ out, int total) {
  for (int i = blockIdx.x * blockDim.x + threadIdx.x; i < total;
       i += gridDim.x * blockDim.x)
    out[i] = (_Float16)in[i];
}

// deconv weights [Ci][Co][4] (f32) -> [parity][tap][Ci][Co] (f16)
// even (parity0): tap0->kk=1, tap1->kk=3 ; odd (parity1): tap0->kk=2, tap1->kk=0
__global__ void xpose_deconv_w(const float* __restrict__ w, _Float16* __restrict__ wt,
                               int Ci, int Co) {
  size_t total = (size_t)4 * Ci * Co;
  for (size_t idx = (size_t)blockIdx.x * blockDim.x + threadIdx.x; idx < total;
       idx += (size_t)gridDim.x * blockDim.x) {
    int co = (int)(idx % Co);
    size_t r = idx / Co;
    int ci = (int)(r % Ci);
    r /= Ci;
    int tap = (int)(r % 2);
    int parity = (int)(r / 2);
    int kk = (parity == 0) ? (tap == 0 ? 1 : 3) : (tap == 0 ? 2 : 0);
    wt[idx] = (_Float16)w[((size_t)ci * Co + co) * 4 + kk];
  }
}

// ---------------------------------------------------------------------------
// conv1: batch [N][128][1024] f32 --(k=5,pad=2 GEMM + bias + maxpool2)-->
//        out1 [N][64][512] f16.   grid(256, 8), block 256 (8 waves).
// B panel (128K x 64N f16 = 16KB) staged via async copies, double-buffered.
// Loops nested t/kb so the A row pointer is a direct select (stays a
// global_load, no scratch-held pointer array).
// ---------------------------------------------------------------------------
__global__ void conv1_kernel(const float* __restrict__ batch,
                             const _Float16* __restrict__ w1t,
                             const float* __restrict__ bias,
                             const float* __restrict__ zpage,
                             _Float16* __restrict__ out1) {
  __shared__ __align__(16) _Float16 smB[2][128 * 64];
  const int lane = threadIdx.x & 31, wave = threadIdx.x >> 5;
  const int tid = threadIdx.x;
  const int n = blockIdx.x;
  const int col0 = blockIdx.y * 64;
  const int l0 = wave * 16;
  const int hi = (lane >> 4) * 8;

  v8f c[4];
#pragma unroll
  for (int i = 0; i < 4; ++i)
#pragma unroll
    for (int j = 0; j < 8; ++j) c[i][j] = 0.f;

  // stage one 128x64 B chunk: chunk = t*8 + kc/128
  auto stage = [&](int buf, int chunk) {
    const int t = chunk >> 3;
    const int kc = (chunk & 7) * 128;
    const _Float16* Bt = w1t + ((size_t)t * CIN + kc) * C1c + col0;
#pragma unroll
    for (int i = 0; i < 4; ++i) {
      int e = tid + i * 256;          // 1024 x 16B transfers
      int r = e >> 3, seg = e & 7;
      async_copy_b128(Bt + (size_t)r * C1c + seg * 8,
                      &smB[buf][r * 64 + seg * 8]);
    }
  };

  stage(0, 0);
  int chunk = 0;
  for (int t = 0; t < 5; ++t) {
    const int lsrc = l0 + (lane & 15) + t - 2;
    const float* arowt = (lsrc >= 0 && lsrc < Ll)
                             ? batch + ((long long)n * Ll + lsrc) * CIN
                             : zpage;
    for (int kb = 0; kb < 8; ++kb, ++chunk) {
      const int cur = chunk & 1;
      wait_asynccnt0();      // our buf[cur] copies landed
      __syncthreads();       // everyone's landed; prev buf consumers done
      if (chunk + 1 < 40) stage(cur ^ 1, chunk + 1);
      const float* ar_ = arowt + kb * 128;
      const _Float16* Bl = smB[cur];
#pragma unroll
      for (int k0 = 0; k0 < 128; k0 += 32) {
        __builtin_prefetch(ar_ + k0 + 128, 0, 1);
        v16h a  = a_frag_f32(ar_, k0, hi);
        v16h b0 = b_frag_lds(Bl, k0, 0, lane);
        v16h b1 = b_frag_lds(Bl, k0, 16, lane);
        v16h b2 = b_frag_lds(Bl, k0, 32, lane);
        v16h b3 = b_frag_lds(Bl, k0, 48, lane);
        c[0] = wmma16(a, b0, c[0]);
        c[1] = wmma16(a, b1, c[1]);
        c[2] = wmma16(a, b2, c[2]);
        c[3] = wmma16(a, b3, c[3]);
      }
      __syncthreads();       // done reading buf[cur] before it is re-staged
    }
  }

  const int phi = (lane >> 4) * 4;
#pragma unroll
  for (int nt = 0; nt < 4; ++nt) {
    int col = col0 + nt * 16 + (lane & 15);
    float bi = bias[col];
#pragma unroll
    for (int i = 0; i < 4; ++i) {
      float v = fmaxf(c[nt][2 * i], c[nt][2 * i + 1]) + bi;
      int p = l0 / 2 + phi + i;
      out1[((size_t)n * 64 + p) * C1c + col] = (_Float16)v;
    }
  }
}

// ---------------------------------------------------------------------------
// conv2: out1 [N][64][512] f16 --(k=5,pad=2 + bias + maxpool2)-->
//        tmp2f [N][32][256] f32  AND  ar[...,0:256] f16.  grid(128, 4).
// ---------------------------------------------------------------------------
__global__ void conv2_kernel(const _Float16* __restrict__ out1,
                             const _Float16* __restrict__ w2t,
                             const float* __restrict__ bias,
                             const float* __restrict__ zpage,
                             float* __restrict__ tmp2f, _Float16* __restrict__ ar) {
  const int lane = threadIdx.x & 31, wave = threadIdx.x >> 5;
  const int m0 = blockIdx.x * 128 + wave * 16;
  const int n = m0 / 64, p0 = m0 % 64;
  const int col0 = blockIdx.y * 64;
  const int hi = (lane >> 4) * 8;
  const _Float16* zp = (const _Float16*)zpage;
  v8f c[4];
#pragma unroll
  for (int i = 0; i < 4; ++i)
#pragma unroll
    for (int j = 0; j < 8; ++j) c[i][j] = 0.f;

  for (int t = 0; t < 5; ++t) {
    int psrc = p0 + (lane & 15) + t - 2;
    const _Float16* arow = (psrc >= 0 && psrc < 64)
                               ? out1 + ((long long)n * 64 + psrc) * C1c
                               : zp;
    const _Float16* Bt = w2t + (size_t)t * C1c * C2c;
    for (int k0 = 0; k0 < C1c; k0 += 32) {
      v16h a  = a_frag_f16(arow, k0, hi);
      v16h b0 = b_frag(Bt, C2c, k0, col0, lane);
      v16h b1 = b_frag(Bt, C2c, k0, col0 + 16, lane);
      v16h b2 = b_frag(Bt, C2c, k0, col0 + 32, lane);
      v16h b3 = b_frag(Bt, C2c, k0, col0 + 48, lane);
      c[0] = wmma16(a, b0, c[0]);
      c[1] = wmma16(a, b1, c[1]);
      c[2] = wmma16(a, b2, c[2]);
      c[3] = wmma16(a, b3, c[3]);
    }
  }
  const int phi = (lane >> 4) * 4;
#pragma unroll
  for (int nt = 0; nt < 4; ++nt) {
    int col = col0 + nt * 16 + (lane & 15);
    float bi = bias[col];
#pragma unroll
    for (int i = 0; i < 4; ++i) {
      float v = fmaxf(c[nt][2 * i], c[nt][2 * i + 1]) + bi;
      int p2 = p0 / 2 + phi + i;
      size_t row = (size_t)n * L4 + p2;
      tmp2f[row * C2c + col] = v;
      ar[row * 1024 + col] = (_Float16)v;  // slice 0 of concat
    }
  }
}

// ---------------------------------------------------------------------------
// generic f16 GEMM + bias -> f32 (qp/kp/ckp projections).  grid(M/128, N/64).
// ---------------------------------------------------------------------------
__global__ void gemm_qkv(const _Float16* __restrict__ A, int lda,
                         const _Float16* __restrict__ Bw, int ldb,
                         const float* __restrict__ bias, float* __restrict__ out,
                         int ldo, int K) {
  const int lane = threadIdx.x & 31, wave = threadIdx.x >> 5;
  const int m0 = blockIdx.x * 128 + wave * 16;
  const int col0 = blockIdx.y * 64;
  const int hi = (lane >> 4) * 8;
  const _Float16* arow = A + (size_t)(m0 + (lane & 15)) * lda;
  v8f c[4];
#pragma unroll
  for (int i = 0; i < 4; ++i)
#pragma unroll
    for (int j = 0; j < 8; ++j) c[i][j] = 0.f;

  for (int k0 = 0; k0 < K; k0 += 32) {
    v16h a  = a_frag_f16(arow, k0, hi);
    v16h b0 = b_frag(Bw, ldb, k0, col0, lane);
    v16h b1 = b_frag(Bw, ldb, k0, col0 + 16, lane);
    v16h b2 = b_frag(Bw, ldb, k0, col0 + 32, lane);
    v16h b3 = b_frag(Bw, ldb, k0, col0 + 48, lane);
    c[0] = wmma16(a, b0, c[0]);
    c[1] = wmma16(a, b1, c[1]);
    c[2] = wmma16(a, b2, c[2]);
    c[3] = wmma16(a, b3, c[3]);
  }
  const int m8 = (lane >> 4) * 8;
#pragma unroll
  for (int nt = 0; nt < 4; ++nt) {
    int col = col0 + nt * 16 + (lane & 15);
    float bi = bias[col];
#pragma unroll
    for (int i = 0; i < 8; ++i)
      out[(size_t)(m0 + m8 + i) * ldo + col] = c[nt][i] + bi;
  }
}

// ---------------------------------------------------------------------------
// deconv (ConvTranspose1d k=4 s=2 p=1) as two k=2 GEMM parity passes.
// A [N][Pin][lda] f16, Wt [2][2][K][ldb] f16, out rows o = 2p+parity.
// grid(M/128, Ncols/64, 2=parity)
// ---------------------------------------------------------------------------
template <typename OutT>
__global__ void deconv_kernel(const _Float16* __restrict__ A, int lda, int Pin,
                              const _Float16* __restrict__ Wt, int K, int ldb,
                              const float* __restrict__ bias,
                              const float* __restrict__ zpage,
                              OutT* __restrict__ out, int ldo) {
  const int lane = threadIdx.x & 31, wave = threadIdx.x >> 5;
  const int parity = blockIdx.z;
  const int m0 = blockIdx.x * 128 + wave * 16;
  const int n = m0 / Pin, p0 = m0 % Pin;
  const int col0 = blockIdx.y * 64;
  const int hi = (lane >> 4) * 8;
  const _Float16* zp = (const _Float16*)zpage;
  v8f c[4];
#pragma unroll
  for (int i = 0; i < 4; ++i)
#pragma unroll
    for (int j = 0; j < 8; ++j) c[i][j] = 0.f;

#pragma unroll
  for (int tap = 0; tap < 2; ++tap) {
    int shift = (tap == 0) ? 0 : (parity ? 1 : -1);
    int psrc = p0 + (lane & 15) + shift;
    const _Float16* arow = (psrc >= 0 && psrc < Pin)
                               ? A + ((long long)n * Pin + psrc) * lda
                               : zp;
    const _Float16* Bt = Wt + (size_t)(parity * 2 + tap) * K * ldb;
    for (int k0 = 0; k0 < K; k0 += 32) {
      v16h a  = a_frag_f16(arow, k0, hi);
      v16h b0 = b_frag(Bt, ldb, k0, col0, lane);
      v16h b1 = b_frag(Bt, ldb, k0, col0 + 16, lane);
      v16h b2 = b_frag(Bt, ldb, k0, col0 + 32, lane);
      v16h b3 = b_frag(Bt, ldb, k0, col0 + 48, lane);
      c[0] = wmma16(a, b0, c[0]);
      c[1] = wmma16(a, b1, c[1]);
      c[2] = wmma16(a, b2, c[2]);
      c[3] = wmma16(a, b3, c[3]);
    }
  }
  const int m8 = (lane >> 4) * 8;
#pragma unroll
  for (int nt = 0; nt < 4; ++nt) {
    int col = col0 + nt * 16 + (lane & 15);
    float bi = bias[col];
#pragma unroll
    for (int i = 0; i < 8; ++i) {
      int p = p0 + m8 + i;
      int o = 2 * p + parity;
      out[((size_t)n * (2 * Pin) + o) * ldo + col] = (OutT)(c[nt][i] + bi);
    }
  }
}

// ---------------------------------------------------------------------------
// self additive attention.  one block per sequence n. block 256.
// writes self_res f16 into ar[...,256:512]
// ---------------------------------------------------------------------------
__global__ void attn_self_kernel(const float* __restrict__ qp,
                                 const float* __restrict__ kp,
                                 const float* __restrict__ tmp2f,
                                 const float* __restrict__ sa_v,
                                 const float* __restrict__ sa_vb,
                                 const int* __restrict__ seg_len,
                                 _Float16* __restrict__ ar) {
  __shared__ float sA[L4 * C2c];        // qp, later tmp2
  __shared__ _Float16 sk[L4 * C2c];     // kp (f16)
  __shared__ float ss[L4 * L4];
  __shared__ float sv[C2c];
  const int n = blockIdx.x, tid = threadIdx.x;
  const float* qn = qp + (size_t)n * L4 * C2c;
  const float* kn = kp + (size_t)n * L4 * C2c;
  for (int i = tid; i < L4 * C2c; i += 256) {
    sA[i] = qn[i];
    sk[i] = (_Float16)kn[i];
  }
  sv[tid] = sa_v[tid];
  __syncthreads();
  const float vb = sa_vb[0];
  const int sl = seg_len[n];
  for (int e = tid; e < L4 * L4; e += 256) {
    int q = e >> 5, k = e & 31;
    float acc = vb;
    const float* qrow = sA + q * C2c;
    const _Float16* krow = sk + k * C2c;
    for (int h = 0; h < C2c; ++h) acc += tanhf(qrow[h] + (float)krow[h]) * sv[h];
    ss[e] = (sl > 4 * k) ? acc : -1e9f;
  }
  __syncthreads();
  const float* tn = tmp2f + (size_t)n * L4 * C2c;
  for (int i = tid; i < L4 * C2c; i += 256) sA[i] = tn[i];  // now tmp2
  if (tid < L4) {  // softmax per query row
    float* row = ss + tid * L4;
    float mx = row[0];
    for (int k = 1; k < L4; ++k) mx = fmaxf(mx, row[k]);
    float sum = 0.f;
    for (int k = 0; k < L4; ++k) { float e = __expf(row[k] - mx); row[k] = e; sum += e; }
    float inv = 1.f / sum;
    for (int k = 0; k < L4; ++k) row[k] *= inv;
  }
  __syncthreads();
  const int ccol = tid;  // 256 channels
  for (int q = 0; q < L4; ++q) {
    float acc = 0.f;
    for (int k = 0; k < L4; ++k) acc += ss[q * L4 + k] * sA[k * C2c + ccol];
    ar[((size_t)n * L4 + q) * 1024 + 256 + ccol] = (_Float16)acc;
  }
}

// ---------------------------------------------------------------------------
// concept attention (both concepts).  one block per n. writes ar[...,512:1024]
// ---------------------------------------------------------------------------
__global__ void attn_concept_kernel(const float* __restrict__ concept1,
                                    const float* __restrict__ concept2,
                                    const float* __restrict__ ca_wq,
                                    const float* __restrict__ ca_bq,
                                    const float* __restrict__ ckp,
                                    const float* __restrict__ ca_v,
                                    const float* __restrict__ ca_vb,
                                    const float* __restrict__ tmp2f,
                                    const int* __restrict__ seg_len,
                                    _Float16* __restrict__ ar) {
  __shared__ float st[L4 * C2c];
  __shared__ float cq[C2c];
  __shared__ float sc[L4];
  __shared__ float sv[C2c];
  const int n = blockIdx.x, tid = threadIdx.x;
  const int b = n >> 5;
  const float* tn = tmp2f + (size_t)n * L4 * C2c;
  for (int i = tid; i < L4 * C2c; i += 256) st[i] = tn[i];
  sv[tid] = ca_v[tid];
  const float vb = ca_vb[0];
  const int sl = seg_len[n];
  for (int cc = 0; cc < 2; ++cc) {
    const float* cp = (cc == 0 ? concept1 : concept2) + (size_t)b * CDd;
    __syncthreads();
    {
      float acc = ca_bq[tid];
      for (int d = 0; d < CDd; ++d) acc += cp[d] * ca_wq[d * C2c + tid];
      cq[tid] = acc;
    }
    __syncthreads();
    if (tid < L4) {
      float acc = vb;
      const float* kr = ckp + ((size_t)n * L4 + tid) * C2c;
      for (int h = 0; h < C2c; ++h) acc += tanhf(cq[h] + kr[h]) * sv[h];
      sc[tid] = (sl > 4 * tid) ? acc : -1e9f;
    }
    __syncthreads();
    if (tid == 0) {
      float mx = sc[0];
      for (int k = 1; k < L4; ++k) mx = fmaxf(mx, sc[k]);
      float sum = 0.f;
      for (int k = 0; k < L4; ++k) { float e = __expf(sc[k] - mx); sc[k] = e; sum += e; }
      float inv = 1.f / sum;
      for (int k = 0; k < L4; ++k) sc[k] *= inv;
    }
    __syncthreads();
    float acc = 0.f;
    for (int k = 0; k < L4; ++k) acc += sc[k] * st[k * C2c + tid];
    _Float16 hv = (_Float16)acc;
    for (int q = 0; q < L4; ++q)
      ar[((size_t)n * L4 + q) * 1024 + 512 + cc * 256 + tid] = hv;
    __syncthreads();
  }
}

// ---------------------------------------------------------------------------
// score folding: v_c[b][d] = sum_j sim1_w[d][j] * (concept_c@sim2_w)[b][j]*mlp_w[j]
// ---------------------------------------------------------------------------
__global__ void score_prep_kernel(const float* __restrict__ c1,
                                  const float* __restrict__ c2,
                                  const float* __restrict__ sim2_w,
                                  const float* __restrict__ sim1_w,
                                  const float* __restrict__ mlp_w,
                                  float* __restrict__ vv) {
  __shared__ float we[2][SIMs];
  const int b = blockIdx.x, j = threadIdx.x;
  for (int cc = 0; cc < 2; ++cc) {
    const float* cp = (cc == 0 ? c1 : c2) + (size_t)b * CDd;
    float q = 0.f;
    for (int d = 0; d < CDd; ++d) q += cp[d] * sim2_w[d * SIMs + j];
    we[cc][j] = q * mlp_w[j];
  }
  __syncthreads();
  for (int cc = 0; cc < 2; ++cc) {
    float acc = 0.f;
    for (int jj = 0; jj < SIMs; ++jj) acc += sim1_w[j * SIMs + jj] * we[cc][jj];
    vv[((size_t)cc * Bb + b) * D2d + j] = acc;
  }
}

// score: out[c][m] = sigmoid(r[m,:]·v_c[b] + mlp_b), m = 0..32767
__global__ void score_kernel(const float* __restrict__ r,
                             const float* __restrict__ vv,
                             const float* __restrict__ mlp_b,
                             float* __restrict__ out) {
  const int m = blockIdx.x * blockDim.x + threadIdx.x;
  if (m >= Nn * Ll) return;
  const int b = m / (Ss * Ll);
  const float* rr = r + (size_t)m * D2d;
  const float* v1 = vv + (size_t)b * D2d;
  const float* v2 = vv + (size_t)(Bb + b) * D2d;
  float a1 = mlp_b[0], a2 = mlp_b[0];
  for (int d = 0; d < D2d; ++d) {
    float x = rr[d];
    a1 += x * v1[d];
    a2 += x * v2[d];
  }
  out[m] = 1.f / (1.f + __expf(-a1));
  out[(size_t)Nn * Ll + m] = 1.f / (1.f + __expf(-a2));
}

// ---------------------------------------------------------------------------
// launch
// ---------------------------------------------------------------------------
extern "C" void kernel_launch(void* const* d_in, const int* in_sizes, int n_in,
                              void* d_out, int out_size, void* d_ws, size_t ws_size,
                              hipStream_t stream) {
  const float* batch    = (const float*)d_in[0];
  const float* concept1 = (const float*)d_in[1];
  const float* concept2 = (const float*)d_in[2];
  const int*   seg_len  = (const int*)d_in[3];
  const float* conv1_w  = (const float*)d_in[4];
  const float* conv1_b  = (const float*)d_in[5];
  const float* conv2_w  = (const float*)d_in[6];
  const float* conv2_b  = (const float*)d_in[7];
  const float* sa_wq    = (const float*)d_in[8];
  const float* sa_bq    = (const float*)d_in[9];
  const float* sa_wk    = (const float*)d_in[10];
  const float* sa_bk    = (const float*)d_in[11];
  const float* sa_v     = (const float*)d_in[12];
  const float* sa_vb    = (const float*)d_in[13];
  const float* ca_wq    = (const float*)d_in[14];
  const float* ca_bq    = (const float*)d_in[15];
  const float* ca_wk    = (const float*)d_in[16];
  const float* ca_bk    = (const float*)d_in[17];
  const float* ca_v     = (const float*)d_in[18];
  const float* ca_vb    = (const float*)d_in[19];
  const float* dc1_w    = (const float*)d_in[20];
  const float* dc1_b    = (const float*)d_in[21];
  const float* dc2_w    = (const float*)d_in[22];
  const float* dc2_b    = (const float*)d_in[23];
  const float* sim1_w   = (const float*)d_in[24];
  const float* sim2_w   = (const float*)d_in[25];
  const float* mlp_w    = (const float*)d_in[26];
  const float* mlp_b    = (const float*)d_in[27];
  float* out = (float*)d_out;

  // workspace carve (~113 MB total)
  char* p = (char*)d_ws;
  auto take = [&](size_t bytes) -> void* {
    void* r = (void*)p;
    p += (bytes + 255) & ~(size_t)255;
    return r;
  };
  _Float16* w1t   = (_Float16*)take((size_t)5 * CIN * C1c * 2);
  _Float16* w2t   = (_Float16*)take((size_t)5 * C1c * C2c * 2);
  _Float16* wqh   = (_Float16*)take((size_t)C2c * C2c * 2);
  _Float16* wkh   = (_Float16*)take((size_t)C2c * C2c * 2);
  _Float16* cwkh  = (_Float16*)take((size_t)C2c * C2c * 2);
  _Float16* dc1t  = (_Float16*)take((size_t)4 * 1024 * D1d * 2);
  _Float16* dc2t  = (_Float16*)take((size_t)4 * D1d * D2d * 2);
  _Float16* out1  = (_Float16*)take((size_t)Nn * 64 * C1c * 2);
  float*    tmp2f = (float*)take((size_t)Nn * L4 * C2c * 4);
  _Float16* ar    = (_Float16*)take((size_t)Nn * L4 * 1024 * 2);
  float*    qp    = (float*)take((size_t)Nn * L4 * C2c * 4);
  float*    kp    = (float*)take((size_t)Nn * L4 * C2c * 4);
  float*    ckp   = (float*)take((size_t)Nn * L4 * C2c * 4);
  _Float16* dc1o  = (_Float16*)take((size_t)Nn * 64 * D1d * 2);
  float*    rbuf  = (float*)take((size_t)Nn * Ll * D2d * 4);
  float*    vv    = (float*)take((size_t)2 * Bb * D2d * 4);
  float*    zpage = (float*)take((size_t)1024 * 4);

  // --- weight prep + zero page ---
  zero_fill<<<4, 256, 0, stream>>>(zpage, 1024);
  xpose_conv_w<<<2048, 256, 0, stream>>>(conv1_w, w1t, C1c, CIN, 5);
  xpose_conv_w<<<1024, 256, 0, stream>>>(conv2_w, w2t, C2c, C1c, 5);
  cvt_f16_kernel<<<256, 256, 0, stream>>>(sa_wq, wqh, C2c * C2c);
  cvt_f16_kernel<<<256, 256, 0, stream>>>(sa_wk, wkh, C2c * C2c);
  cvt_f16_kernel<<<256, 256, 0, stream>>>(ca_wk, cwkh, C2c * C2c);
  xpose_deconv_w<<<2048, 256, 0, stream>>>(dc1_w, dc1t, 1024, D1d);
  xpose_deconv_w<<<512, 256, 0, stream>>>(dc2_w, dc2t, D1d, D2d);

  // --- conv stack (WMMA implicit GEMM, fused maxpool, async B staging) ---
  conv1_kernel<<<dim3(Nn, C1c / 64), 256, 0, stream>>>(batch, w1t, conv1_b, zpage,
                                                       out1);
  conv2_kernel<<<dim3(Nn * 64 / 128, C2c / 64), 256, 0, stream>>>(
      out1, w2t, conv2_b, zpage, tmp2f, ar);

  // --- attention projections (WMMA GEMM) ---
  dim3 gqkv(Nn * L4 / 128, C2c / 64);
  gemm_qkv<<<gqkv, 256, 0, stream>>>(ar, 1024, wqh, C2c, sa_bq, qp, C2c, C2c);
  gemm_qkv<<<gqkv, 256, 0, stream>>>(ar, 1024, wkh, C2c, sa_bk, kp, C2c, C2c);
  gemm_qkv<<<gqkv, 256, 0, stream>>>(ar, 1024, cwkh, C2c, ca_bk, ckp, C2c, C2c);

  // --- attention (VALU/LDS) ---
  attn_self_kernel<<<Nn, 256, 0, stream>>>(qp, kp, tmp2f, sa_v, sa_vb, seg_len, ar);
  attn_concept_kernel<<<Nn, 256, 0, stream>>>(concept1, concept2, ca_wq, ca_bq, ckp,
                                              ca_v, ca_vb, tmp2f, seg_len, ar);

  // --- deconvs as parity-split WMMA GEMMs ---
  deconv_kernel<_Float16><<<dim3(Nn * L4 / 128, D1d / 64, 2), 256, 0, stream>>>(
      ar, 1024, L4, dc1t, 1024, D1d, dc1_b, zpage, dc1o, D1d);
  deconv_kernel<float><<<dim3(Nn * 64 / 128, D2d / 64, 2), 256, 0, stream>>>(
      dc1o, D1d, 64, dc2t, D1d, D2d, dc2_b, zpage, rbuf, D2d);

  // --- folded similarity/score ---
  score_prep_kernel<<<Bb, SIMs, 0, stream>>>(concept1, concept2, sim2_w, sim1_w,
                                             mlp_w, vv);
  score_kernel<<<(Nn * Ll + 255) / 256, 256, 0, stream>>>(rbuf, vv, mlp_b, out);
}
